// SelfAttentionHead_84293028151412
// MI455X (gfx1250) — matrix operand order repaired and checked by reading, not compiled, not run on hardware
//
#include <hip/hip_runtime.h>
#include <stdint.h>

typedef unsigned short u16;
typedef __attribute__((ext_vector_type(16))) __bf16 v16bf;
typedef __attribute__((ext_vector_type(8)))  float  v8f;

union Frag {
  v16bf  v;
  uint4  q[2];
  __bf16 e[16];
  u16    s[16];
};

static __device__ __forceinline__ u16 f2bf(float f) {
  union { __bf16 b; u16 s; } u;
  u.b = (__bf16)f;
  return u.s;
}

#define B_N 4
#define T_N 4096
#define C_N 1024
#define H_N 64

#define WT_ELEMS  (H_N * C_N)                 // 65536 elems per transposed weight
#define QKV_ELEMS ((size_t)B_N * T_N * H_N)   // per tensor

// ---------------------------------------------------------------------------
// Kernel 1: Wq/Wk/Wv [C][H] fp32  ->  W^T [H][C] bf16 in workspace
// ---------------------------------------------------------------------------
__global__ void prep_weights(const float* __restrict__ Wq, const float* __restrict__ Wk,
                             const float* __restrict__ Wv, u16* __restrict__ ws) {
  const int i = blockIdx.x * blockDim.x + threadIdx.x;
  if (i >= 3 * WT_ELEMS) return;
  const int m   = i >> 16;              // which weight (WT_ELEMS == 65536)
  const int rem = i & (WT_ELEMS - 1);
  const int h   = rem >> 10;            // / C_N
  const int c   = rem & (C_N - 1);
  const float* W = (m == 0) ? Wq : (m == 1) ? Wk : Wv;
  ws[(size_t)m * WT_ELEMS + (size_t)h * C_N + c] = f2bf(W[(size_t)c * H_N + h]);
}

// ---------------------------------------------------------------------------
// Kernel 2: projections. One wave (32 threads) computes 32 rows of q,k,v.
// A-frag: x fp32 loaded directly (two 32B runs / lane), cvt to bf16.
// B-frag: two 16B runs / lane from W^T bf16.
// Outputs: q,k bf16 [B][T][H] row-major;  v bf16 transposed [B][H][T].
// ---------------------------------------------------------------------------
__global__ __launch_bounds__(32)
void proj_kernel(const float* __restrict__ x, u16* __restrict__ ws) {
  const int lane = threadIdx.x;
  const int hi   = lane >> 4;       // which 16-lane half
  const int l16  = lane & 15;
  const int bidx = blockIdx.x;      // over B*(T/32)
  const int b    = bidx / (T_N / 32);
  const int rowc = bidx % (T_N / 32);
  const int row0 = rowc * 32;

  const u16* wt0 = ws;
  u16* qo = ws + 3 * WT_ELEMS;
  u16* ko = qo + QKV_ELEMS;
  u16* vo = ko + QKV_ELEMS;         // transposed [B][H][T]

  v8f acc[3][4][2];
  #pragma unroll
  for (int w = 0; w < 3; ++w)
    #pragma unroll
    for (int nt = 0; nt < 4; ++nt)
      #pragma unroll
      for (int t = 0; t < 2; ++t) acc[w][nt][t] = (v8f){};

  const float* xb = x + ((size_t)b * T_N + row0) * C_N;

  for (int kk = 0; kk < C_N; kk += 32) {
    Frag a[2];
    #pragma unroll
    for (int t = 0; t < 2; ++t) {
      // A layout 16x32 bf16: lanes 0-15 K = kk+{0..7,16..23}; lanes 16-31 +8
      const float* xr = xb + (size_t)(t * 16 + l16) * C_N + kk + (hi ? 8 : 0);
      float4 tv[4];
      tv[0] = *(const float4*)(xr);
      tv[1] = *(const float4*)(xr + 4);
      tv[2] = *(const float4*)(xr + 16);
      tv[3] = *(const float4*)(xr + 20);
      const float* tf = (const float*)tv;
      #pragma unroll
      for (int e2 = 0; e2 < 16; ++e2) a[t].e[e2] = (__bf16)tf[e2];
    }
    #pragma unroll
    for (int w = 0; w < 3; ++w) {
      #pragma unroll
      for (int nt = 0; nt < 4; ++nt) {
        Frag bf;
        // B layout 32x16: lane N = nt*16+l16; K = kk+{0..7,16..23} (lo half) / +8 (hi half)
        const u16* wp = wt0 + (size_t)w * WT_ELEMS +
                        (size_t)(nt * 16 + l16) * C_N + kk + (hi ? 8 : 0);
        bf.q[0] = *(const uint4*)(wp);
        bf.q[1] = *(const uint4*)(wp + 16);
        #pragma unroll
        for (int t = 0; t < 2; ++t)
          acc[w][nt][t] = __builtin_amdgcn_wmma_f32_16x16x32_bf16(
              false, a[t].v, false, bf.v, (short)0, acc[w][nt][t], false, false);
      }
    }
  }

  // Epilogue: C/D layout -> lane l holds col nt*16+l16, row r + 8*hi per VGPR r
  #pragma unroll
  for (int t = 0; t < 2; ++t)
    #pragma unroll
    for (int nt = 0; nt < 4; ++nt) {
      const int h = nt * 16 + l16;
      #pragma unroll
      for (int r = 0; r < 8; ++r) {
        const int trow = row0 + t * 16 + r + hi * 8;
        const size_t off = ((size_t)b * T_N + trow) * H_N + h;
        qo[off] = f2bf(acc[0][nt][t][r]);
        ko[off] = f2bf(acc[1][nt][t][r]);
      }
      // v transposed: per lane the 8 row values are contiguous t's -> one b128
      union { uint4 u; u16 s[8]; } pk;
      #pragma unroll
      for (int r = 0; r < 8; ++r) pk.s[r] = f2bf(acc[2][nt][t][r]);
      const int tbase = row0 + t * 16 + hi * 8;
      *(uint4*)(vo + ((size_t)b * H_N + h) * T_N + tbase) = pk.u;
    }
}

// ---------------------------------------------------------------------------
// Kernel 3: flash attention. One wave handles 32 query rows (two 16-row tiles),
// streaming causal key blocks of 32 from L2. WMMA for S=QK^T and O+=PV; online
// softmax with lane-shuffle row reductions; P transposed C/D->A via wave-local
// LDS (single-wave workgroup => DS in-order, barrier is free).
// ---------------------------------------------------------------------------
__global__ __launch_bounds__(32)
void attn_kernel(const u16* __restrict__ ws, float* __restrict__ out) {
  __align__(16) __shared__ u16 Pbuf[2][16][40];   // padded rows (80B stride)

  const int lane = threadIdx.x;
  const int hi   = lane >> 4;
  const int l16  = lane & 15;
  const int bidx = blockIdx.x;
  const int b    = bidx / (T_N / 32);
  const int rowc = bidx % (T_N / 32);
  const int row0 = rowc * 32;

  const u16* qg  = ws + 3 * WT_ELEMS;
  const u16* kg  = qg + QKV_ELEMS;
  const u16* vtg = kg + QKV_ELEMS;
  const u16* qb  = qg  + (size_t)b * T_N * H_N;
  const u16* kb  = kg  + (size_t)b * T_N * H_N;
  const u16* vtb = vtg + (size_t)b * H_N * T_N;

  // Q A-fragments stay resident: 2 row tiles x 2 h-frags (K = h 0..31, 32..63)
  Frag qa[2][2];
  #pragma unroll
  for (int t = 0; t < 2; ++t)
    #pragma unroll
    for (int f = 0; f < 2; ++f) {
      const u16* p = qb + (size_t)(row0 + t * 16 + l16) * H_N + f * 32 + (hi ? 8 : 0);
      qa[t][f].q[0] = *(const uint4*)(p);        // K = base..base+7
      qa[t][f].q[1] = *(const uint4*)(p + 16);   // K = base+16..base+23
    }

  v8f   o[2][4];
  float mrow[2][8], lrow[2][8];
  #pragma unroll
  for (int t = 0; t < 2; ++t) {
    #pragma unroll
    for (int nt = 0; nt < 4; ++nt) o[t][nt] = (v8f){};
    #pragma unroll
    for (int r = 0; r < 8; ++r) { mrow[t][r] = -__builtin_inff(); lrow[t][r] = 0.0f; }
  }

  const float scale = 0.03125f;   // 1/sqrt(C) = 1/32  (reference scales by C**0.5)
  const int nblk = rowc + 1;      // causal: key blocks of 32 up through the diagonal

  for (int kbi = 0; kbi < nblk; ++kbi) {
    const int  kb0  = kbi * 32;
    const bool last = (kbi == nblk - 1);

    // K^T B-frags: [n-tile of 16 keys][h-frag]; lane N = key, K = h {base..base+7, base+16..base+23}
    Frag kf[2][2];
    #pragma unroll
    for (int nt = 0; nt < 2; ++nt)
      #pragma unroll
      for (int f = 0; f < 2; ++f) {
        const u16* p = kb + (size_t)(kb0 + nt * 16 + l16) * H_N + f * 32 + (hi ? 8 : 0);
        kf[nt][f].q[0] = *(const uint4*)(p);
        kf[nt][f].q[1] = *(const uint4*)(p + 16);
      }
    // V B-frags from transposed v: lane N = h col, K = keys {base..base+7, base+16..base+23}
    Frag vf[4];
    #pragma unroll
    for (int nt = 0; nt < 4; ++nt) {
      const u16* p = vtb + (size_t)(nt * 16 + l16) * T_N + kb0 + (hi ? 8 : 0);
      vf[nt].q[0] = *(const uint4*)(p);
      vf[nt].q[1] = *(const uint4*)(p + 16);
    }

    #pragma unroll
    for (int t = 0; t < 2; ++t) {
      v8f s0 = (v8f){}, s1 = (v8f){};
      s0 = __builtin_amdgcn_wmma_f32_16x16x32_bf16(false, qa[t][0].v, false, kf[0][0].v, (short)0, s0, false, false);
      s0 = __builtin_amdgcn_wmma_f32_16x16x32_bf16(false, qa[t][1].v, false, kf[0][1].v, (short)0, s0, false, false);
      s1 = __builtin_amdgcn_wmma_f32_16x16x32_bf16(false, qa[t][0].v, false, kf[1][0].v, (short)0, s1, false, false);
      s1 = __builtin_amdgcn_wmma_f32_16x16x32_bf16(false, qa[t][1].v, false, kf[1][1].v, (short)0, s1, false, false);
      #pragma unroll
      for (int r = 0; r < 8; ++r) { s0[r] *= scale; s1[r] *= scale; }

      if (last) {   // causal mask only touches the diagonal key block
        #pragma unroll
        for (int r = 0; r < 8; ++r) {
          const int row = row0 + t * 16 + r + hi * 8;
          if (kb0 + l16      > row) s0[r] = -__builtin_inff();
          if (kb0 + 16 + l16 > row) s1[r] = -__builtin_inff();
        }
      }

      // row max across 32 keys (cols live across the 16 lanes of each half)
      float mn[8];
      #pragma unroll
      for (int r = 0; r < 8; ++r) mn[r] = fmaxf(s0[r], s1[r]);
      #pragma unroll
      for (int msk = 1; msk <= 8; msk <<= 1)
        #pragma unroll
        for (int r = 0; r < 8; ++r) mn[r] = fmaxf(mn[r], __shfl_xor(mn[r], msk, 32));

      float al[8];
      #pragma unroll
      for (int r = 0; r < 8; ++r) {
        mn[r]      = fmaxf(mn[r], mrow[t][r]);
        al[r]      = __expf(mrow[t][r] - mn[r]);
        mrow[t][r] = mn[r];
      }
      #pragma unroll
      for (int r = 0; r < 8; ++r) {
        s0[r] = __expf(s0[r] - mn[r]);
        s1[r] = __expf(s1[r] - mn[r]);
      }
      float rs[8];
      #pragma unroll
      for (int r = 0; r < 8; ++r) rs[r] = s0[r] + s1[r];
      #pragma unroll
      for (int msk = 1; msk <= 8; msk <<= 1)
        #pragma unroll
        for (int r = 0; r < 8; ++r) rs[r] += __shfl_xor(rs[r], msk, 32);
      #pragma unroll
      for (int r = 0; r < 8; ++r) lrow[t][r] = lrow[t][r] * al[r] + rs[r];

      // rescale running O, stash P (bf16) in LDS in row-major [16][32]
      #pragma unroll
      for (int nt = 0; nt < 4; ++nt)
        #pragma unroll
        for (int r = 0; r < 8; ++r) o[t][nt][r] *= al[r];
      #pragma unroll
      for (int r = 0; r < 8; ++r) {
        Pbuf[t][r + hi * 8][l16]      = f2bf(s0[r]);
        Pbuf[t][r + hi * 8][16 + l16] = f2bf(s1[r]);
      }
    }

    __syncthreads();   // single-wave WG: free; orders LDS store->load
    #pragma unroll
    for (int t = 0; t < 2; ++t) {
      Frag pf;   // A layout: lane row = l16; K = keys {c0..c0+7, c0+16..c0+23}
      const int c0 = hi ? 8 : 0;
      pf.q[0] = *(const uint4*)(&Pbuf[t][l16][c0]);
      pf.q[1] = *(const uint4*)(&Pbuf[t][l16][c0 + 16]);
      #pragma unroll
      for (int nt = 0; nt < 4; ++nt)
        o[t][nt] = __builtin_amdgcn_wmma_f32_16x16x32_bf16(
            false, pf.v, false, vf[nt].v, (short)0, o[t][nt], false, false);
    }
    __syncthreads();
  }

  // normalize + store fp32 output [B][T][H]
  #pragma unroll
  for (int t = 0; t < 2; ++t)
    #pragma unroll
    for (int nt = 0; nt < 4; ++nt) {
      const int h = nt * 16 + l16;
      #pragma unroll
      for (int r = 0; r < 8; ++r) {
        const int trow = row0 + t * 16 + r + hi * 8;
        out[((size_t)b * T_N + trow) * H_N + h] = o[t][nt][r] / lrow[t][r];
      }
    }
}

// ---------------------------------------------------------------------------
extern "C" void kernel_launch(void* const* d_in, const int* in_sizes, int n_in,
                              void* d_out, int out_size, void* d_ws, size_t ws_size,
                              hipStream_t stream) {
  (void)in_sizes; (void)n_in; (void)out_size; (void)ws_size;
  const float* x  = (const float*)d_in[0];
  const float* Wq = (const float*)d_in[1];
  const float* Wk = (const float*)d_in[2];
  const float* Wv = (const float*)d_in[3];
  u16*   ws  = (u16*)d_ws;
  float* out = (float*)d_out;

  prep_weights<<<(3 * WT_ELEMS + 255) / 256, 256, 0, stream>>>(Wq, Wk, Wv, ws);
  proj_kernel<<<B_N * (T_N / 32), 32, 0, stream>>>(x, ws);
  attn_kernel<<<B_N * (T_N / 32), 32, 0, stream>>>(ws, out);
}
